// HDGMC_27273042330335
// MI455X (gfx1250) — matrix operand order repaired and checked by reading, not compiled, not run on hardware
//
#include <hip/hip_runtime.h>
#include <hip/hip_bf16.h>
#include <math.h>

// ---------------- problem constants (match reference) ----------------
#define B_    8
#define N_    1024
#define DEG_  16
#define CIN_  128
#define COUT_ 256
#define RIN_  32
#define ROUT_ 32
#define KSEL  32
#define STEPS_ 2
#define BN_   (B_*N_)          // 8192
#define E_    (B_*N_*DEG_)     // 131072

typedef __bf16 bf16;
typedef __attribute__((ext_vector_type(16))) __bf16 v16bf;
typedef __attribute__((ext_vector_type(8)))  __bf16 v8bf;
typedef __attribute__((ext_vector_type(8)))  float  v8f;
typedef __attribute__((ext_vector_type(4)))  unsigned u32x4;
typedef __attribute__((ext_vector_type(8)))  int      i32x8;
typedef __attribute__((ext_vector_type(4)))  int      i32x4;

__device__ __forceinline__ float wred_sum(float v){
  #pragma unroll
  for(int o=16;o>0;o>>=1) v += __shfl_xor(v,o);
  return v;
}
__device__ __forceinline__ float wred_max(float v){
  #pragma unroll
  for(int o=16;o>0;o>>=1) v = fmaxf(v,__shfl_xor(v,o));
  return v;
}
__device__ __forceinline__ unsigned hash_u32(unsigned x){
  x ^= x >> 16; x *= 0x7feb352du; x ^= x >> 15; x *= 0x846ca68bu; x ^= x >> 16; return x;
}

// TDM: 1-D contiguous copy global -> LDS, nbytes must be a multiple of 8.
// Descriptor layout per CDNA5 ISA ch.8 (group0: count/lds/global/type, group1: data_size + dims).
__device__ __forceinline__ void tdm_copy_1d(const void* gptr, void* lptr, unsigned nbytes){
  unsigned long long ga = (unsigned long long)gptr;
  unsigned lds = (unsigned)(unsigned long long)lptr;   // generic addr low 32 bits == LDS offset
  unsigned ne8 = nbytes >> 3;                          // 8-byte elements
  u32x4 g0;
  g0[0] = 1u;                                          // count=1 (valid), user descriptor
  g0[1] = lds;                                         // lds_addr
  g0[2] = (unsigned)ga;                                // global_addr[31:0]
  g0[3] = (unsigned)((ga >> 32) & 0x01FFFFFFu) | (2u << 30);  // global_addr[56:32] | type=2
  i32x8 g1;
  g1[0] = (int)(3u << 16);                             // data_size = 3 (8 bytes)
  g1[1] = (int)((ne8 & 0xFFFFu) << 16);                // tensor_dim0[15:0]  (bits 63:48)
  g1[2] = (int)((ne8 >> 16) & 0xFFFFu);                // tensor_dim0[31:16] (bits 79:64)
  g1[3] = (int)((ne8 & 0xFFFFu) << 16);                // tile_dim0 (bits 127:112)
  g1[4] = 0; g1[5] = 0; g1[6] = 0; g1[7] = 0;          // tile_dim1/2=0, strides=0 (1-D)
  i32x4 z4; z4[0]=0; z4[1]=0; z4[2]=0; z4[3]=0;
#if defined(__clang_major__) && (__clang_major__ >= 23)
  i32x8 z8; z8[0]=0; z8[1]=0; z8[2]=0; z8[3]=0; z8[4]=0; z8[5]=0; z8[6]=0; z8[7]=0;
  __builtin_amdgcn_tensor_load_to_lds(g0, g1, z4, z4, z8, 0);
#else
  __builtin_amdgcn_tensor_load_to_lds(g0, g1, z4, z4, 0);
#endif
}

// ---------------- utility kernels ----------------
__global__ void k_zero(float* __restrict__ p, int n){
  for(int i = blockIdx.x*blockDim.x + threadIdx.x; i < n; i += gridDim.x*blockDim.x) p[i] = 0.f;
}

// agg[dst,c] += x[src,c] over all edges.  edges: [2,E] int64, C = 1<<cshift.
__global__ void k_scatter_add(const float* __restrict__ x, const long long* __restrict__ edges,
                              float* __restrict__ agg, int cshift, int mask0){
  const int C = 1 << cshift;
  const int total = E_ << cshift;
  for(int i = blockIdx.x*blockDim.x + threadIdx.x; i < total; i += gridDim.x*blockDim.x){
    int e = i >> cshift, c = i & (C-1);
    if(mask0 && c == 0) continue;                  // logmap0 zeroes the time coordinate
    int src = ((int)edges[e])      & (BN_-1);
    int dst = ((int)edges[E_ + e]) & (BN_-1);
    atomicAdd(&agg[(size_t)dst*C + c], x[(size_t)src*C + c]);
  }
}

__global__ void k_add_cvt(const float* __restrict__ x, const float* __restrict__ agg,
                          bf16* __restrict__ y, int n){
  for(int i = blockIdx.x*blockDim.x + threadIdx.x; i < n; i += gridDim.x*blockDim.x)
    y[i] = (bf16)(x[i] + agg[i]);
}

// wt[n*K+k] = (bf16) w[k*Ncols+n]   (pre-transpose weights so B-fragments load contiguously)
__global__ void k_cvt_tr(const float* __restrict__ w, bf16* __restrict__ wt, int K, int Ncols){
  int total = K*Ncols;
  for(int i = blockIdx.x*blockDim.x + threadIdx.x; i < total; i += gridDim.x*blockDim.x){
    int k = i / Ncols, n = i % Ncols;
    wt[(size_t)n*K + k] = (bf16)w[i];
  }
}

// psi2 input: (0, r[1:]) + agg  ->  bf16
__global__ void k_hin_cvt(const float* __restrict__ r, const float* __restrict__ agg,
                          bf16* __restrict__ y, int n){
  for(int i = blockIdx.x*blockDim.x + threadIdx.x; i < n; i += gridDim.x*blockDim.x){
    int c = i & (RIN_-1);
    float v = (c == 0) ? 0.f : (r[i] + agg[i]);
    y[i] = (bf16)v;
  }
}

// ------- WMMA bf16 GEMM with TDM-staged A:  C[M,Nc] = relu?( A[M,K] * Bt[Nc,K]^T ) -------
// Block owns `mt` consecutive M-tiles and ALL Nc/16 column tiles; A block (mt*16 rows x K,
// contiguous, mt*16*K == 2048 bf16 here) is DMA'd to LDS once by the Tensor Data Mover,
// then all 8 waves build A-fragments from LDS while B streams through L2.
__global__ void __launch_bounds__(256)
k_gemm_tdm(const bf16* __restrict__ A, const bf16* __restrict__ Bt,
           float* __restrict__ C, int M, int Nc, int K, int relu, int mt){
  __shared__ bf16 Atile[2048];                         // 4 KB staged A block
  int wid = threadIdx.x >> 5, lane = threadIdx.x & 31;
  int half = lane >> 4, l16 = lane & 15;
  int tilesN = Nc >> 4;
  int tm0 = blockIdx.x * mt;                           // first M-tile of this block
  if(wid == 0){
    tdm_copy_1d(A + (size_t)tm0*16*K, Atile, (unsigned)(mt*16*K)*2u);
    __builtin_amdgcn_s_wait_tensorcnt(0);
  }
  __syncthreads();

  int nwt = mt * tilesN;                               // wave-tiles in this block
  for(int wt = wid; wt < nwt; wt += 8){
    int tml = wt / tilesN, tn = wt % tilesN;
    const bf16* arow = Atile + (tml*16 + l16)*K;       // LDS
    const bf16* brow = Bt + (size_t)(tn*16 + l16)*K;   // global (L2-resident weights)
    v8f acc;
    #pragma unroll
    for(int r=0;r<8;r++) acc[r] = 0.f;
    for(int kb = 0; kb < K; kb += 32){
      if(kb + 32 < K) __builtin_prefetch(brow + kb + 32, 0, 1);   // global_prefetch_b8
      // A fragment (ISA 16-bit A 16x32 layout): lanes0-15 K={kb..kb+7,kb+16..kb+23}, lanes16-31 +8
      v8bf alo = *(const v8bf*)(arow + kb +      half*8);
      v8bf ahi = *(const v8bf*)(arow + kb + 16 + half*8);
      // B fragment: same K order as A: lanes0-15 K={kb..kb+7,kb+16..kb+23}, lanes16-31 +8
      v8bf blo = *(const v8bf*)(brow + kb +      half*8);
      v8bf bhi = *(const v8bf*)(brow + kb + 16 + half*8);
      v16bf a, b;
      #pragma unroll
      for(int j=0;j<8;j++){ a[j]=alo[j]; a[j+8]=ahi[j]; b[j]=blo[j]; b[j+8]=bhi[j]; }
      acc = __builtin_amdgcn_wmma_f32_16x16x32_bf16(false, a, false, b, (short)0, acc, false, false);
    }
    #pragma unroll
    for(int r=0;r<8;r++){
      float v = acc[r];
      if(relu) v = fmaxf(v, 0.f);
      C[(size_t)((tm0 + tml)*16 + r + 8*half)*Nc + tn*16 + l16] = v;  // D: VGPR r -> M=r(+8)
    }
  }
}

// ---------------- psi1 tail: h = proj(expmap0(t)) row-wise (C = 256), in place, + bf16 copy ----
__global__ void k_expmap_proj(float* __restrict__ h, bf16* __restrict__ hb, int rows){
  int wid = threadIdx.x >> 5, lane = threadIdx.x & 31;
  int row = blockIdx.x*(blockDim.x >> 5) + wid;
  if(row >= rows) return;
  float* p = h + (size_t)row*COUT_;
  float vals[COUT_/32];
  float ss = 0.f;
  #pragma unroll
  for(int q=0;q<COUT_/32;q++){
    int c = q*32 + lane;
    float t = p[c];
    vals[q] = t;
    if(c > 0) ss += t*t;
  }
  ss = wred_sum(ss);
  float n = fmaxf(sqrtf(ss), 1e-8f);
  float f = sinhf(n)/n;                      // spatial scale
  float h0 = sqrtf(1.f + f*f*ss);            // proj time coord (== cosh(n))
  #pragma unroll
  for(int q=0;q<COUT_/32;q++){
    int c = q*32 + lane;
    float o = (c == 0) ? h0 : f*vals[q];
    p[c] = o;
    hb[(size_t)row*COUT_ + c] = (bf16)o;
  }
}

// ---------------- fused Lorentz similarity (WMMA bf16) + per-row top-32 in LDS --------------
__global__ void __launch_bounds__(256)
k_sim_topk(const bf16* __restrict__ hbs, const bf16* __restrict__ hbt,
           const float* __restrict__ hs, const float* __restrict__ ht,
           float* __restrict__ Shat, int* __restrict__ Sidx){
  __shared__ float Srow[16*N_];                 // 64 KB score block: 16 source rows x 1024 cols
  int b  = blockIdx.x / (N_/16);
  int it = blockIdx.x % (N_/16);
  int i0 = it*16;
  int wid = threadIdx.x >> 5, lane = threadIdx.x & 31;
  int half = lane >> 4, l16 = lane & 15;

  // hoist the 8 A-fragments (K = 256) for this source tile
  const bf16* arow = hbs + (size_t)(b*N_ + i0 + l16)*COUT_;
  v16bf afrag[8];
  #pragma unroll
  for(int kk=0; kk<8; kk++){
    int kb = kk*32;
    v8bf lo = *(const v8bf*)(arow + kb +      half*8);
    v8bf hi = *(const v8bf*)(arow + kb + 16 + half*8);
    #pragma unroll
    for(int j=0;j<8;j++){ afrag[kk][j]=lo[j]; afrag[kk][j+8]=hi[j]; }
  }
  float h0s[8];
  #pragma unroll
  for(int r=0;r<8;r++) h0s[r] = hs[(size_t)(b*N_ + i0 + r + 8*half)*COUT_];

  for(int jt = wid; jt < N_/16; jt += 8){       // 8 waves stripe the 64 column tiles
    const bf16* brow = hbt + (size_t)(b*N_ + jt*16 + l16)*COUT_;   // h_t row-major == Bt
    v8f acc;
    #pragma unroll
    for(int r=0;r<8;r++) acc[r] = 0.f;
    #pragma unroll
    for(int kk=0; kk<8; kk++){
      int kb = kk*32;
      v8bf lo = *(const v8bf*)(brow + kb +      half*8);
      v8bf hi = *(const v8bf*)(brow + kb + 16 + half*8);
      v16bf bfrag;
      #pragma unroll
      for(int j=0;j<8;j++){ bfrag[j]=lo[j]; bfrag[j+8]=hi[j]; }
      acc = __builtin_amdgcn_wmma_f32_16x16x32_bf16(false, afrag[kk], false, bfrag,
                                                    (short)0, acc, false, false);
    }
    float h0t = ht[(size_t)(b*N_ + jt*16 + l16)*COUT_];
    #pragma unroll
    for(int r=0;r<8;r++)
      Srow[(r + 8*half)*N_ + jt*16 + l16] = acc[r] - 2.f*h0s[r]*h0t;   // Lorentz inner product
  }
  __syncthreads();

  // sorted top-32 per row (iterative wave argmax, tie -> lowest index, like lax.top_k)
  #pragma unroll
  for(int rr=0; rr<2; rr++){
    int row = wid*2 + rr;
    float* S = Srow + row*N_;
    int grow = b*N_ + i0 + row;
    for(int t=0; t<KSEL; t++){
      float best = -3.4e38f; int bj = 0;
      for(int q=0; q<32; q++){
        int j = q*32 + lane;                      // stride-32 scan: conflict-free banks
        float v = S[j];
        if(v > best){ best = v; bj = j; }
      }
      #pragma unroll
      for(int o=16;o>0;o>>=1){
        float v2 = __shfl_xor(best, o);
        int   j2 = __shfl_xor(bj, o);
        if(v2 > best || (v2 == best && j2 < bj)){ best = v2; bj = j2; }
      }
      if(lane == (bj & 31)) S[bj] = -3.4e38f;
      if(lane == 0){
        Shat[(size_t)grow*KSEL + t] = best;
        Sidx[(size_t)grow*KSEL + t] = b*N_ + bj;  // global segment index
      }
    }
  }
}

// ---------------- row softmax over K=32 ----------------
__global__ void k_softmax32(const float* __restrict__ Shat, float* __restrict__ out, int rows){
  int wid = threadIdx.x >> 5, lane = threadIdx.x & 31;
  int row = blockIdx.x*(blockDim.x >> 5) + wid;
  if(row >= rows) return;
  float v = Shat[(size_t)row*KSEL + lane];
  float m = wred_max(v);
  float e = __expf(v - m);
  float s = wred_sum(e);
  out[(size_t)row*KSEL + lane] = e/s;
}

// ---------------- r_s ~ N(0,1)/10, deterministic per step ----------------
__global__ void k_rng(float* __restrict__ r, unsigned step, int n){
  for(int i = blockIdx.x*blockDim.x + threadIdx.x; i < n; i += gridDim.x*blockDim.x){
    unsigned s1 = hash_u32(2u*i     + step*0x9E3779B9u + 0x1234567u);
    unsigned s2 = hash_u32(2u*i + 1 + step*0x85EBCA6Bu + 0x89ABCDFu);
    float u1 = (float)(s1 >> 8)*(1.f/16777216.f) + 1e-7f;
    float u2 = (float)(s2 >> 8)*(1.f/16777216.f);
    r[i] = 0.1f * sqrtf(-2.f*__logf(u1)) * __cosf(6.28318530718f*u2);
  }
}

// ---------------- r_t[seg[bn,k]] += r_s[bn] * softmax(Shat[bn])[k] ----------------
__global__ void k_scatter_soft(const float* __restrict__ Shat, const int* __restrict__ Sidx,
                               const float* __restrict__ rs, float* __restrict__ rt){
  int wid = threadIdx.x >> 5, lane = threadIdx.x & 31;
  int bn = blockIdx.x*(blockDim.x >> 5) + wid;
  if(bn >= BN_) return;
  float v = Shat[(size_t)bn*KSEL + lane];
  float m = wred_max(v);
  float e = __expf(v - m);
  float s = wred_sum(e);
  float sm = e/s;                                   // lane acts as k
  int gidx = Sidx[(size_t)bn*KSEL + lane];
  float rv = rs[(size_t)bn*RIN_ + lane];            // lane acts as channel c
  for(int k=0;k<KSEL;k++){
    float sk = __shfl(sm, k);
    int   ik = __shfl(gidx, k);
    atomicAdd(&rt[(size_t)ik*RIN_ + lane], rv*sk);
  }
}

// ---------------- MLP step: S_hat += (relu(D@Wm1+bm1)@Wm2+bm2); D = (0, z_s[1:]-z_t[1:]) ----
__global__ void k_mstep(const float* __restrict__ zs, const float* __restrict__ zt,
                        const int* __restrict__ Sidx,
                        const float* __restrict__ Wm1, const float* __restrict__ bm1,
                        const float* __restrict__ Wm2, const float* __restrict__ bm2,
                        float* __restrict__ Shat){
  __shared__ float w1s[ROUT_*10], w2s[10], b1s[10];
  int tid = threadIdx.x;
  if(tid < ROUT_*10) w1s[tid] = Wm1[tid];
  if(tid < 10){ w2s[tid] = Wm2[tid]; b1s[tid] = bm1[tid]; }
  __syncthreads();
  float b2 = bm2[0];
  int gid = blockIdx.x*blockDim.x + tid;
  if(gid >= BN_*KSEL) return;
  int bn = gid >> 5;
  int gidx = Sidx[gid];
  float hid[10];
  #pragma unroll
  for(int j=0;j<10;j++) hid[j] = b1s[j];
  for(int c=1;c<ROUT_;c++){                          // D[0] == 0 exactly
    float d = zs[(size_t)bn*ROUT_ + c] - zt[(size_t)gidx*ROUT_ + c];
    #pragma unroll
    for(int j=0;j<10;j++) hid[j] = fmaf(d, w1s[c*10 + j], hid[j]);
  }
  float mo = b2;
  #pragma unroll
  for(int j=0;j<10;j++) mo += fmaxf(hid[j], 0.f)*w2s[j];
  Shat[gid] += mo;
}

// ================================= host launcher =================================
static inline int nblk(long n, int t){ long b = (n + t - 1)/t; return (int)(b > 65535 ? 65535 : b); }

extern "C" void kernel_launch(void* const* d_in, const int* in_sizes, int n_in,
                              void* d_out, int out_size, void* d_ws, size_t ws_size,
                              hipStream_t stream){
  const float*     x_s = (const float*)d_in[0];
  const long long* e_s = (const long long*)d_in[1];
  const float*     x_t = (const float*)d_in[3];
  const long long* e_t = (const long long*)d_in[4];
  const float*     W1  = (const float*)d_in[6];
  const float*     W2  = (const float*)d_in[7];
  const float*     Wm1 = (const float*)d_in[8];
  const float*     bm1 = (const float*)d_in[9];
  const float*     Wm2 = (const float*)d_in[10];
  const float*     bm2 = (const float*)d_in[11];
  float* out = (float*)d_out;

  char* w = (char*)d_ws;
  float* agg  = (float*)(w);                         //  4 MB  BN*CIN
  float* h_s  = (float*)(w + (4ull<<20));            //  8 MB  BN*COUT
  float* h_t  = (float*)(w + (12ull<<20));           //  8 MB
  float* Shat = (float*)(w + (20ull<<20));           //  1 MB  BN*K
  int*   Sidx = (int*)  (w + (21ull<<20));           //  1 MB
  float* r_s  = (float*)(w + (22ull<<20));           //  1 MB  BN*RIN
  float* r_t  = (float*)(w + (23ull<<20));           //  1 MB
  float* z_s  = (float*)(w + (24ull<<20));           //  1 MB
  float* z_t  = (float*)(w + (25ull<<20));           //  1 MB
  float* agg2 = (float*)(w + (26ull<<20));           //  1 MB
  bf16*  ybf  = (bf16*) (w + (27ull<<20));           //  2 MB  BN*CIN bf16
  bf16*  hb_s = (bf16*) (w + (29ull<<20));           //  4 MB  BN*COUT bf16
  bf16*  hb_t = (bf16*) (w + (33ull<<20));           //  4 MB
  bf16*  W1t  = (bf16*) (w + (37ull<<20));           // 64 KB  COUT x CIN
  bf16*  W2t  = (bf16*) (w + (37ull<<20) + 65536);   //  2 KB  ROUT x RIN
  bf16*  zbf  = (bf16*) (w + (38ull<<20));           // .5 MB  BN*RIN bf16

  const int T = 256;

  // weight pre-transpose + bf16 conversion (once)
  k_cvt_tr<<<nblk(CIN_*COUT_,T), T, 0, stream>>>(W1, W1t, CIN_, COUT_);
  k_cvt_tr<<<nblk(RIN_*ROUT_,T), T, 0, stream>>>(W2, W2t, RIN_, ROUT_);

  // ---- psi1 for source, then target (reusing agg/ybf) ----
  const float* xs[2]      = { x_s, x_t };
  const long long* es[2]  = { e_s, e_t };
  float* hh[2]            = { h_s, h_t };
  bf16*  hbb[2]           = { hb_s, hb_t };
  for(int p=0; p<2; p++){
    k_zero<<<nblk(BN_*CIN_,T), T, 0, stream>>>(agg, BN_*CIN_);
    k_scatter_add<<<8192, T, 0, stream>>>(xs[p], es[p], agg, 7 /*CIN=128*/, 0);
    k_add_cvt<<<nblk(BN_*CIN_,T), T, 0, stream>>>(xs[p], agg, ybf, BN_*CIN_);
    // mt=1: block = one 16-row A block (16x128 bf16 = 4KB via TDM) x 16 column tiles
    k_gemm_tdm<<<BN_/16, T, 0, stream>>>(ybf, W1t, hh[p], BN_, COUT_, CIN_, 1, 1);
    k_expmap_proj<<<BN_/8, T, 0, stream>>>(hh[p], hbb[p], BN_);
  }

  // ---- fused similarity + top-32 ----
  k_sim_topk<<<B_*(N_/16), T, 0, stream>>>(hb_s, hb_t, h_s, h_t, Shat, Sidx);

  // ---- S_0 ----
  k_softmax32<<<BN_/8, T, 0, stream>>>(Shat, out, BN_);

  // ---- refinement steps ----
  for(int step=0; step<STEPS_; step++){
    k_rng<<<nblk(BN_*RIN_,T), T, 0, stream>>>(r_s, (unsigned)step, BN_*RIN_);
    k_zero<<<nblk(BN_*RIN_,T), T, 0, stream>>>(r_t, BN_*RIN_);
    k_scatter_soft<<<BN_/8, T, 0, stream>>>(Shat, Sidx, r_s, r_t);

    // psi2 (source / target); logmap0(proj(expmap0(.))) collapses to (0, .[1:]) at c=1
    const float* rr[2] = { r_s, r_t };
    float* zz[2]       = { z_s, z_t };
    for(int p=0; p<2; p++){
      k_zero<<<nblk(BN_*RIN_,T), T, 0, stream>>>(agg2, BN_*RIN_);
      k_scatter_add<<<4096, T, 0, stream>>>(rr[p], es[p], agg2, 5 /*RIN=32*/, 1);
      k_hin_cvt<<<nblk(BN_*RIN_,T), T, 0, stream>>>(rr[p], agg2, zbf, BN_*RIN_);
      // mt=4: block = 64-row A block (64x32 bf16 = 4KB via TDM) x 2 column tiles
      k_gemm_tdm<<<BN_/64, T, 0, stream>>>(zbf, W2t, zz[p], BN_, ROUT_, RIN_, 1, 4);
    }
    k_mstep<<<nblk((long)BN_*KSEL,T), T, 0, stream>>>(z_s, z_t, Sidx, Wm1, bm1, Wm2, bm2, Shat);
  }

  // ---- S_L ----
  k_softmax32<<<BN_/8, T, 0, stream>>>(Shat, out + (size_t)BN_*KSEL, BN_);
}
